// Attention_45440753992365
// MI455X (gfx1250) — matrix operand order repaired and checked by reading, not compiled, not run on hardware
//
#include <hip/hip_runtime.h>

typedef __attribute__((ext_vector_type(16))) _Float16 v16h;
typedef __attribute__((ext_vector_type(8)))  _Float16 v8h;
typedef __attribute__((ext_vector_type(8)))  float    v8f;

#define B_     4
#define NLAT   512
#define NCTX   4096
#define NTOT   4608          // NLAT + NCTX (queries prepended to context)
#define DIM    512
#define H_     8
#define DH     64
#define DINNER 512
#define KDIM   512

#define ASTR   40            // lds row stride (f16): 80B = 20 banks, 16B aligned
#define KSTR   72            // 144B = 36 banks, 16B aligned
#define VSTR   40
#define PSTR   40
#define CHUNK  32

__device__ __forceinline__ v16h cat16(v8h a, v8h b) {
    return __builtin_shufflevector(a, b, 0,1,2,3,4,5,6,7,8,9,10,11,12,13,14,15);
}

// per-lane 16B async copy global -> LDS (ASYNCcnt tracked)
__device__ __forceinline__ void async_b128(unsigned lds_off, const void* gptr) {
    asm volatile("global_load_async_to_lds_b128 %0, %1, off"
                 :: "v"(lds_off), "v"((unsigned long long)(uintptr_t)gptr)
                 : "memory");
}

// ---------------------------------------------------------------------------
// Generic GEMM: C[M,N] = A[M,K] * W[N,K]^T   (K = 512)
// mode 0: A=f32 x             -> f16 qbuf [b,h,n,64], scaled by d^-1/2
// mode 1: A=f32 concat(x,ctx) -> f16 kbuf [b,h,m,64], vtbuf [b,h,d,m] (V^T)
// mode 2: A=f16 obuf          -> f32 d_out [b*n, 512]
// ---------------------------------------------------------------------------
__global__ __launch_bounds__(128)
void proj_kernel(const float* __restrict__ A1, const float* __restrict__ A2,
                 const float* __restrict__ W, const _Float16* __restrict__ A16,
                 _Float16* __restrict__ out_a, _Float16* __restrict__ out_b,
                 float* __restrict__ out_f32, int mode)
{
    __shared__ _Float16 lA[64 * ASTR];
    __shared__ _Float16 lB[64 * ASTR];

    const int tid  = threadIdx.x;
    const int wave = tid >> 5;
    const int lane = tid & 31;
    const int r0   = blockIdx.x * 64;
    const int c0   = blockIdx.y * 64;

    v8f acc[4] = {};

    for (int k0 = 0; k0 < KDIM; k0 += 32) {
        // ---- A chunk (64 rows x 32) -> f16 LDS
        {
            int row  = tid >> 1;
            int colb = (tid & 1) * 16;
            int gr   = r0 + row;
            _Float16* dst = &lA[row * ASTR + colb];
            if (mode == 2) {
                const _Float16* s16 = A16 + (size_t)gr * KDIM + k0 + colb;
                *(v8h*)dst       = *(const v8h*)s16;
                *(v8h*)(dst + 8) = *(const v8h*)(s16 + 8);
            } else {
                const float* src;
                if (mode == 1) {
                    int bb = gr / NTOT;
                    int i  = gr - bb * NTOT;
                    src = (i < NLAT)
                        ? (A1 + ((size_t)bb * NLAT + i) * KDIM + k0 + colb)
                        : (A2 + ((size_t)bb * NCTX + (i - NLAT)) * KDIM + k0 + colb);
                } else {
                    src = A1 + (size_t)gr * KDIM + k0 + colb;
                }
                v8h h0, h1;
#pragma unroll
                for (int j = 0; j < 8; j += 4) {
                    float4 f = *(const float4*)(src + j);
                    h0[j+0] = (_Float16)f.x; h0[j+1] = (_Float16)f.y;
                    h0[j+2] = (_Float16)f.z; h0[j+3] = (_Float16)f.w;
                }
#pragma unroll
                for (int j = 0; j < 8; j += 4) {
                    float4 f = *(const float4*)(src + 8 + j);
                    h1[j+0] = (_Float16)f.x; h1[j+1] = (_Float16)f.y;
                    h1[j+2] = (_Float16)f.z; h1[j+3] = (_Float16)f.w;
                }
                *(v8h*)dst       = h0;
                *(v8h*)(dst + 8) = h1;
            }
        }
        // ---- W chunk (64 out-features x 32 K)
        {
            int row  = tid >> 1;
            int colb = (tid & 1) * 16;
            const float* src = W + (size_t)(c0 + row) * KDIM + k0 + colb;
            _Float16* dst = &lB[row * ASTR + colb];
            v8h h0, h1;
#pragma unroll
            for (int j = 0; j < 8; j += 4) {
                float4 f = *(const float4*)(src + j);
                h0[j+0] = (_Float16)f.x; h0[j+1] = (_Float16)f.y;
                h0[j+2] = (_Float16)f.z; h0[j+3] = (_Float16)f.w;
            }
#pragma unroll
            for (int j = 0; j < 8; j += 4) {
                float4 f = *(const float4*)(src + 8 + j);
                h1[j+0] = (_Float16)f.x; h1[j+1] = (_Float16)f.y;
                h1[j+2] = (_Float16)f.z; h1[j+3] = (_Float16)f.w;
            }
            *(v8h*)dst       = h0;
            *(v8h*)(dst + 8) = h1;
        }
        __syncthreads();

        // ---- A fragment 16x32 (ISA layout): m=lane&15, k=(i>>3)*16+(lane>>4)*8+(i&7)
        const _Float16* ar = &lA[(wave * 16 + (lane & 15)) * ASTR + (lane >> 4) * 8];
        v16h afrag = cat16(*(const v8h*)ar, *(const v8h*)(ar + 16));

#pragma unroll
        for (int nt = 0; nt < 4; ++nt) {
            const _Float16* br = &lB[(nt * 16 + (lane & 15)) * ASTR + (lane >> 4) * 8];
            v16h bfrag = cat16(*(const v8h*)br, *(const v8h*)(br + 16));
            acc[nt] = __builtin_amdgcn_wmma_f32_16x16x32_f16(
                false, afrag, false, bfrag, (short)0, acc[nt], false, false);
        }
        __syncthreads();
    }

    // ---- epilogue: D layout row = j + (lane>>4)*8, col = lane&15
    const int half = lane >> 4;
    const int nidx = lane & 15;
#pragma unroll
    for (int nt = 0; nt < 4; ++nt) {
        int ncol = c0 + nt * 16 + nidx;
        int rbase = r0 + wave * 16 + half * 8;
        if (mode == 1 && ncol >= DINNER) {
            // V^T output: lane's 8 rows are 8 consecutive tokens at fixed d
            int bb = rbase / NTOT, mtok0 = rbase - bb * NTOT;
            int c = ncol - DINNER, hh = c >> 6, d = c & 63;
            v8h vv;
#pragma unroll
            for (int j = 0; j < 8; ++j) vv[j] = (_Float16)acc[nt][j];
            *(v8h*)(out_b + (((size_t)bb * H_ + hh) * DH + d) * NTOT + mtok0) = vv;
        } else {
#pragma unroll
            for (int j = 0; j < 8; ++j) {
                int r = rbase + j;
                float val = acc[nt][j];
                if (mode == 0) {
                    int bb = r >> 9, ntok = r & (NLAT - 1);
                    int hh = ncol >> 6, d = ncol & 63;
                    out_a[(((size_t)bb * H_ + hh) * NLAT + ntok) * DH + d] =
                        (_Float16)(val * 0.125f);
                } else if (mode == 1) {
                    int bb = r / NTOT, mtok = r - bb * NTOT;
                    int hh = ncol >> 6, d = ncol & 63;
                    out_a[(((size_t)bb * H_ + hh) * NTOT + mtok) * DH + d] = (_Float16)val;
                } else {
                    out_f32[(size_t)r * DIM + ncol] = val;
                }
            }
        }
    }
}

// ---------------------------------------------------------------------------
// Fused flash attention per (b,h): out = softmax(q k^T) v
// grid (NLAT/64, H, B), 128 threads; wave owns 16 q rows x 64 dims.
// K/V chunks streamed via global_load_async_to_lds_b128, double buffered.
// ---------------------------------------------------------------------------
__global__ __launch_bounds__(128)
void attn_kernel(const _Float16* __restrict__ qbuf, const _Float16* __restrict__ kbuf,
                 const _Float16* __restrict__ vtbuf, _Float16* __restrict__ obuf)
{
    __shared__ _Float16 lK[2][CHUNK * KSTR];   // K chunk, row-major [m][d]
    __shared__ _Float16 lV[2][DH * VSTR];      // V^T chunk, [d][m]
    __shared__ _Float16 lP[4][16 * PSTR];      // per-wave P relayout tile

    const int tid  = threadIdx.x;
    const int wave = tid >> 5;
    const int lane = tid & 31;
    const int b = blockIdx.z, h = blockIdx.y;
    const int n0 = blockIdx.x * 64 + wave * 16;

    const _Float16* qb  = qbuf  + ((size_t)b * H_ + h) * NLAT * DH;
    const _Float16* kpb = kbuf  + ((size_t)b * H_ + h) * NTOT * DH;
    const _Float16* vtb = vtbuf + ((size_t)b * H_ + h) * (size_t)DH * NTOT;

    // issue async loads for one 32-row chunk into buffer `buf`
    auto issue = [&](int m0, int buf) {
#pragma unroll
        for (int rr = 0; rr < 2; ++rr) {
            int blk = rr * 128 + tid;                 // 8-f16 (16B) block index
            int krow = blk >> 3, kc = (blk & 7) * 8;  // K: 32 rows x 64
            async_b128((unsigned)(uintptr_t)&lK[buf][krow * KSTR + kc],
                       kpb + (size_t)(m0 + krow) * DH + kc);
            int vd = blk >> 2, vc = (blk & 3) * 8;    // V^T: 64 rows x 32
            async_b128((unsigned)(uintptr_t)&lV[buf][vd * VSTR + vc],
                       vtb + (size_t)vd * NTOT + m0 + vc);
        }
    };

    // q fragments (16x64 = two 16x32), kept in registers
    v16h qf[2];
    {
        const _Float16* qr = qb + (size_t)(n0 + (lane & 15)) * DH + (lane >> 4) * 8;
        qf[0] = cat16(*(const v8h*)qr,        *(const v8h*)(qr + 16));
        qf[1] = cat16(*(const v8h*)(qr + 32), *(const v8h*)(qr + 48));
    }

    v8f   acc[4] = {};
    float rm[8], rl[8];
#pragma unroll
    for (int j = 0; j < 8; ++j) { rm[j] = -1e30f; rl[j] = 0.f; }

    issue(0, 0);
    for (int it = 0; it < NTOT / CHUNK; ++it) {
        const int cur = it & 1;
        asm volatile("s_wait_asynccnt 0x0" ::: "memory");
        __syncthreads();                               // chunk visible to all waves
        if (it + 1 < NTOT / CHUNK) issue((it + 1) * CHUNK, cur ^ 1);

        // ---- scores S = q k^T for 32 ctx rows (two 16-col tiles)
        v8f s[2];
#pragma unroll
        for (int nt = 0; nt < 2; ++nt) {
            const _Float16* kr =
                &lK[cur][(nt * 16 + (lane & 15)) * KSTR + (lane >> 4) * 8];
            v16h bf0 = cat16(*(const v8h*)kr,        *(const v8h*)(kr + 16));
            v16h bf1 = cat16(*(const v8h*)(kr + 32), *(const v8h*)(kr + 48));
            v8f z = {};
            z = __builtin_amdgcn_wmma_f32_16x16x32_f16(false, qf[0], false, bf0, (short)0, z, false, false);
            z = __builtin_amdgcn_wmma_f32_16x16x32_f16(false, qf[1], false, bf1, (short)0, z, false, false);
            s[nt] = z;
        }

        // ---- online softmax (rows: vgpr j + (lane>>4)*8; reduce 16 lanes)
        float mnew[8], corr[8];
#pragma unroll
        for (int j = 0; j < 8; ++j) {
            float mx = fmaxf(s[0][j], s[1][j]);
            mx = fmaxf(mx, __shfl_xor(mx, 1, 32));
            mx = fmaxf(mx, __shfl_xor(mx, 2, 32));
            mx = fmaxf(mx, __shfl_xor(mx, 4, 32));
            mx = fmaxf(mx, __shfl_xor(mx, 8, 32));
            mnew[j] = fmaxf(rm[j], mx);
            corr[j] = __expf(rm[j] - mnew[j]);
            rm[j]   = mnew[j];
        }
#pragma unroll
        for (int j = 0; j < 8; ++j) {
            float p0 = __expf(s[0][j] - mnew[j]);
            float p1 = __expf(s[1][j] - mnew[j]);
            s[0][j] = p0; s[1][j] = p1;
            float sum = p0 + p1;
            sum += __shfl_xor(sum, 1, 32);
            sum += __shfl_xor(sum, 2, 32);
            sum += __shfl_xor(sum, 4, 32);
            sum += __shfl_xor(sum, 8, 32);
            rl[j] = rl[j] * corr[j] + sum;
        }

        // ---- P: D-fragment -> LDS -> A-fragment (per-wave tile)
        {
            _Float16* pp = lP[wave];
            int n = lane & 15, hlf = lane >> 4;
#pragma unroll
            for (int j = 0; j < 8; ++j) {
                int m = hlf * 8 + j;
                pp[m * PSTR + n]      = (_Float16)s[0][j];
                pp[m * PSTR + 16 + n] = (_Float16)s[1][j];
            }
        }
        asm volatile("s_wait_dscnt 0x0" ::: "memory");  // intra-wave LDS RAW fence

#pragma unroll
        for (int nt = 0; nt < 4; ++nt)
#pragma unroll
            for (int j = 0; j < 8; ++j) acc[nt][j] *= corr[j];

        // ---- acc += P (16x32) x V (32x64); V^T layout makes B reads contiguous
        const _Float16* pr = lP[wave] + (lane & 15) * PSTR + (lane >> 4) * 8;
        v16h pf = cat16(*(const v8h*)pr, *(const v8h*)(pr + 16));
#pragma unroll
        for (int nt = 0; nt < 4; ++nt) {
            const _Float16* vr =
                &lV[cur][(nt * 16 + (lane & 15)) * VSTR + (lane >> 4) * 8];
            v16h vf = cat16(*(const v8h*)vr, *(const v8h*)(vr + 16));
            acc[nt] = __builtin_amdgcn_wmma_f32_16x16x32_f16(
                false, pf, false, vf, (short)0, acc[nt], false, false);
        }
    }

    // ---- epilogue: divide by row sums, write merged-head f16 [b, n, h*64]
    {
        int hlf = lane >> 4, n = lane & 15;
#pragma unroll
        for (int j = 0; j < 8; ++j) {
            float inv = 1.0f / rl[j];
            int row = n0 + hlf * 8 + j;
            _Float16* orow = obuf + ((size_t)b * NLAT + row) * DINNER + h * DH;
#pragma unroll
            for (int nt = 0; nt < 4; ++nt)
                orow[nt * 16 + n] = (_Float16)(acc[nt][j] * inv);
        }
    }
}

// ---------------------------------------------------------------------------
extern "C" void kernel_launch(void* const* d_in, const int* in_sizes, int n_in,
                              void* d_out, int out_size, void* d_ws, size_t ws_size,
                              hipStream_t stream)
{
    const float* x    = (const float*)d_in[0];
    const float* ctx  = (const float*)d_in[1];
    const float* Wq   = (const float*)d_in[2];
    const float* Wkv  = (const float*)d_in[3];
    const float* Wout = (const float*)d_in[4];
    float* out = (float*)d_out;

    _Float16* q  = (_Float16*)d_ws;                        //  2 MB
    _Float16* k  = q  + (size_t)B_ * H_ * NLAT * DH;       // 18.9 MB
    _Float16* vt = k  + (size_t)B_ * H_ * NTOT * DH;       // 18.9 MB (V^T [b,h,d,m])
    _Float16* o  = vt + (size_t)B_ * H_ * NTOT * DH;       //  2 MB

    // q = x @ Wq^T (pre-scaled by d^-1/2) -> [b,h,n,64] f16
    proj_kernel<<<dim3((B_ * NLAT) / 64, DINNER / 64), 128, 0, stream>>>(
        x, nullptr, Wq, nullptr, q, nullptr, nullptr, 0);
    // kv = concat(x,ctx) @ Wkv^T -> k [b,h,m,64], vt [b,h,d,m] f16
    proj_kernel<<<dim3((B_ * NTOT) / 64, (2 * DINNER) / 64), 128, 0, stream>>>(
        x, ctx, Wkv, nullptr, k, vt, nullptr, 1);
    // fused flash attention -> o [b,n,h*64] f16
    attn_kernel<<<dim3(NLAT / 64, H_, B_), 128, 0, stream>>>(q, k, vt, o);
    // out = o @ Wout^T -> f32 [b,n,512]
    proj_kernel<<<dim3((B_ * NLAT) / 64, DIM / 64), 128, 0, stream>>>(
        nullptr, nullptr, Wout, o, nullptr, nullptr, out, 2);
}
